// LSTMFeatureExtractor_57337813402255
// MI455X (gfx1250) — matrix operand order repaired and checked by reading, not compiled, not run on hardware
//
#include <hip/hip_runtime.h>
#include <hip/hip_bf16.h>

typedef __attribute__((ext_vector_type(16))) _Float16 v16h;
typedef __attribute__((ext_vector_type(8)))  _Float16 v8h;
typedef __attribute__((ext_vector_type(8)))  float    v8f;

#define T_LEN  128
#define D_IN   32
#define H_DIM  64
#define F_DIM  128
#define LN_EPS 1e-5f

// Dynamic LDS layout (bytes). Total 132096 B (< 320KB WGP LDS, 2 blocks/WGP).
#define OFF_WC0   0        // [256][96]  f16 packed [Wih0 | Whh0]  = 49152 B
#define OFF_WC1   49152    // [256][128] f16 packed [Wih1 | Whh1]  = 65536 B
#define OFF_H0    114688   // double-buffered h0 tile: 2*16*64 f16 = 4096 B
#define OFF_H1    118784   // double-buffered h1 tile               = 4096 B
#define OFF_FEATS 122880   // [16][128] f32                         = 8192 B
#define OFF_PART  131072   // [16][8][2] f32 partial sums           = 1024 B
#define SMEM_BYTES 132096

// ---- branch-free activations on the CDNA5 TRANS pipe ----
// gfx1250 has V_TANH_F32; use the builtin if this clang exposes it,
// else fall back to a branch-free v_exp/v_rcp formulation.
#if defined(__has_builtin)
#if __has_builtin(__builtin_amdgcn_tanhf)
#define HAS_NATIVE_TANH 1
#endif
#endif

__device__ __forceinline__ float tanh_(float x) {
#ifdef HAS_NATIVE_TANH
    return __builtin_amdgcn_tanhf(x);               // v_tanh_f32
#else
    float e = __expf(2.0f * x);                     // v_exp_f32 (scaled)
    return 1.0f - 2.0f * __builtin_amdgcn_rcpf(e + 1.0f);  // v_rcp_f32
#endif
}

__device__ __forceinline__ float sigmoid_(float x) {
#ifdef HAS_NATIVE_TANH
    return fmaf(tanh_(0.5f * x), 0.5f, 0.5f);       // v_mul + v_tanh + v_fma
#else
    return __builtin_amdgcn_rcpf(1.0f + __expf(-x));
#endif
}

// CDNA5 16-bit A-fragment (16x32, MxK) from an LDS-resident 16x64 f16 tile.
// Lane l (l<16): row=l, K = {colBase+0..15}
// Lane l (l>=16): row=l-16, K = {colBase+16..31}
__device__ __forceinline__ v16h load_a_frag(const _Float16* hb, int colBase, int ln, int lh) {
    const _Float16* p = hb + ln * H_DIM + colBase + lh * 16;
    v8h lo = *(const v8h*)p;
    v8h hi = *(const v8h*)(p + 8);
    v16h a;
#pragma unroll
    for (int i = 0; i < 8; ++i) { a[i] = lo[i]; a[i + 8] = hi[i]; }
    return a;
}

__device__ __forceinline__ v8f wmma_f16(v16h a, v16h b, v8f c) {
    return __builtin_amdgcn_wmma_f32_16x16x32_f16(false, a, false, b, (short)0, c, false, false);
}

__global__ __launch_bounds__(128, 1)
void lstm_fused_kernel(const float* __restrict__ x,
                       const float* __restrict__ Wih0, const float* __restrict__ Whh0,
                       const float* __restrict__ bih0, const float* __restrict__ bhh0,
                       const float* __restrict__ Wih1, const float* __restrict__ Whh1,
                       const float* __restrict__ bih1, const float* __restrict__ bhh1,
                       const float* __restrict__ Wp,  const float* __restrict__ bp,
                       const float* __restrict__ gamma, const float* __restrict__ beta,
                       float* __restrict__ out)
{
    extern __shared__ __align__(16) char smem[];
    _Float16* wc0  = (_Float16*)(smem + OFF_WC0);
    _Float16* wc1  = (_Float16*)(smem + OFF_WC1);
    _Float16* h0s  = (_Float16*)(smem + OFF_H0);
    _Float16* h1s  = (_Float16*)(smem + OFF_H1);
    float*    feats = (float*)(smem + OFF_FEATS);
    float*    part  = (float*)(smem + OFF_PART);

    const int tid  = threadIdx.x;
    const int w    = tid >> 5;      // wave 0..3 -> hidden-column slice 16w..16w+15
    const int lane = tid & 31;
    const int ln   = lane & 15;
    const int lh   = lane >> 4;
    const int rowBase = blockIdx.x * 16;

    // ---- Stage f16 weights into LDS: wc0 rows = [Wih0 | Whh0] (K=96),
    //      wc1 rows = [Wih1 | Whh1] (K=128). One f32 read + cvt per element.
    for (int i = tid; i < 256 * 96; i += 128) {
        int g = i / 96, k = i - g * 96;
        float v = (k < D_IN) ? Wih0[g * D_IN + k] : Whh0[g * H_DIM + (k - D_IN)];
        wc0[i] = (_Float16)v;
    }
    for (int i = tid; i < 256 * 128; i += 128) {
        int g = i >> 7, k = i & 127;
        float v = (k < H_DIM) ? Wih1[(g << 6) + k] : Whh1[(g << 6) + (k - H_DIM)];
        wc1[i] = (_Float16)v;
    }
    for (int i = tid; i < 2 * 16 * H_DIM; i += 128) {
        h0s[i] = (_Float16)0.0f;
        h1s[i] = (_Float16)0.0f;
    }

    // combined biases for this wave's 4 gate tiles (N = lane&15 within tile)
    float b0v[4], b1v[4];
#pragma unroll
    for (int g4 = 0; g4 < 4; ++g4) {
        int idx = (w + g4 * 4) * 16 + ln;
        b0v[g4] = bih0[idx] + bhh0[idx];
        b1v[g4] = bih1[idx] + bhh1[idx];
    }

    v8f c0v = {};   // c-state layer 0, C/D accumulator layout (16 rows x 16 cols)
    v8f c1v = {};   // c-state layer 1
    __syncthreads();

    const float* xrow = x + (size_t)(rowBase + ln) * T_LEN * D_IN;
    int p = 0;
    for (int t = 0; t < T_LEN; ++t) {
        const int q = p ^ 1;

        // ================= layer 0:  g = [x_t | h0] @ Wcat0^T + b =================
        v16h a0;   // K-chunk 0: x_t columns 0..31 (f32 -> f16 on the fly)
        {
            const float* xp = xrow + t * D_IN + lh * 16;
#pragma unroll
            for (int i = 0; i < 8; ++i) { a0[i] = (_Float16)xp[i]; a0[i + 8] = (_Float16)xp[8 + i]; }
            if (t + 1 < T_LEN) __builtin_prefetch(xp + D_IN, 0, 0);
        }
        v16h a1 = load_a_frag(h0s + p * (16 * H_DIM), 0,  ln, lh);  // K 32..63 = h0[0..31]
        v16h a2 = load_a_frag(h0s + p * (16 * H_DIM), 32, ln, lh);  // K 64..95 = h0[32..63]

        v8f acc[4];
#pragma unroll
        for (int g4 = 0; g4 < 4; ++g4) {
            const _Float16* wrow = wc0 + ((w + g4 * 4) * 16 + ln) * 96 + lh * 16;
            v8f a = {};
#pragma unroll
            for (int e = 0; e < 8; ++e) a[e] = b0v[g4];
            a = wmma_f16(a0, *(const v16h*)(wrow),      a);
            a = wmma_f16(a1, *(const v16h*)(wrow + 32), a);
            a = wmma_f16(a2, *(const v16h*)(wrow + 64), a);
            acc[g4] = a;
        }
#pragma unroll
        for (int e = 0; e < 8; ++e) {
            float iv = sigmoid_(acc[0][e]);
            float fv = sigmoid_(acc[1][e]);
            float gv = tanh_(acc[2][e]);
            float ov = sigmoid_(acc[3][e]);
            float cv = fv * c0v[e] + iv * gv;
            c0v[e] = cv;
            h0s[q * (16 * H_DIM) + (e + lh * 8) * H_DIM + w * 16 + ln] = (_Float16)(ov * tanh_(cv));
        }
        __syncthreads();

        // ================= layer 1:  g = [h0_t | h1] @ Wcat1^T + b =================
        v16h u0 = load_a_frag(h0s + q * (16 * H_DIM), 0,  ln, lh);
        v16h u1 = load_a_frag(h0s + q * (16 * H_DIM), 32, ln, lh);
        v16h u2 = load_a_frag(h1s + p * (16 * H_DIM), 0,  ln, lh);
        v16h u3 = load_a_frag(h1s + p * (16 * H_DIM), 32, ln, lh);
#pragma unroll
        for (int g4 = 0; g4 < 4; ++g4) {
            const _Float16* wrow = wc1 + ((w + g4 * 4) * 16 + ln) * 128 + lh * 16;
            v8f a = {};
#pragma unroll
            for (int e = 0; e < 8; ++e) a[e] = b1v[g4];
            a = wmma_f16(u0, *(const v16h*)(wrow),      a);
            a = wmma_f16(u1, *(const v16h*)(wrow + 32), a);
            a = wmma_f16(u2, *(const v16h*)(wrow + 64), a);
            a = wmma_f16(u3, *(const v16h*)(wrow + 96), a);
            acc[g4] = a;
        }
#pragma unroll
        for (int e = 0; e < 8; ++e) {
            float iv = sigmoid_(acc[0][e]);
            float fv = sigmoid_(acc[1][e]);
            float gv = tanh_(acc[2][e]);
            float ov = sigmoid_(acc[3][e]);
            float cv = fv * c1v[e] + iv * gv;
            c1v[e] = cv;
            h1s[q * (16 * H_DIM) + (e + lh * 8) * H_DIM + w * 16 + ln] = (_Float16)(ov * tanh_(cv));
        }
        __syncthreads();
        p = q;
    }

    // ============ projection (h1_T @ Wp^T + bp) + LayerNorm + tanh ============
    v16h f0 = load_a_frag(h1s + p * (16 * H_DIM), 0,  ln, lh);
    v16h f1 = load_a_frag(h1s + p * (16 * H_DIM), 32, ln, lh);
#pragma unroll
    for (int j = 0; j < 2; ++j) {
        int n = w + j * 4;                      // feature tile: cols 16n..16n+15
        const float* wpr = Wp + (n * 16 + ln) * H_DIM;
        v16h bw0, bw1;
#pragma unroll
        for (int i = 0; i < 16; ++i) {
            bw0[i] = (_Float16)wpr[lh * 16 + i];
            bw1[i] = (_Float16)wpr[32 + lh * 16 + i];
        }
        float bb = bp[n * 16 + ln];
        v8f a = {};
#pragma unroll
        for (int e = 0; e < 8; ++e) a[e] = bb;
        a = wmma_f16(f0, bw0, a);
        a = wmma_f16(f1, bw1, a);
#pragma unroll
        for (int e = 0; e < 8; ++e) feats[(e + lh * 8) * F_DIM + n * 16 + ln] = a[e];
    }
    __syncthreads();

    {
        const int r  = tid & 15;    // batch row in tile
        const int ch = tid >> 4;    // 0..7 -> 16-col chunk
        float s = 0.f, ss = 0.f;
#pragma unroll
        for (int i = 0; i < 16; ++i) {
            float v = feats[r * F_DIM + ch * 16 + i];
            s += v; ss += v * v;
        }
        part[(r * 8 + ch) * 2 + 0] = s;
        part[(r * 8 + ch) * 2 + 1] = ss;
        __syncthreads();
        float S = 0.f, SS = 0.f;
#pragma unroll
        for (int i = 0; i < 8; ++i) { S += part[(r * 8 + i) * 2]; SS += part[(r * 8 + i) * 2 + 1]; }
        float mu   = S * (1.0f / F_DIM);
        float var  = SS * (1.0f / F_DIM) - mu * mu;
        float rstd = rsqrtf(var + LN_EPS);
        float* orow = out + (size_t)(rowBase + r) * F_DIM;
#pragma unroll
        for (int i = 0; i < 16; ++i) {
            int col = ch * 16 + i;
            float v = (feats[r * F_DIM + col] - mu) * rstd * gamma[col] + beta[col];
            orow[col] = tanh_(v);
        }
    }
}

extern "C" void kernel_launch(void* const* d_in, const int* in_sizes, int n_in,
                              void* d_out, int out_size, void* d_ws, size_t ws_size,
                              hipStream_t stream)
{
    const float* x     = (const float*)d_in[0];
    const float* Wih0  = (const float*)d_in[1];
    const float* Whh0  = (const float*)d_in[2];
    const float* bih0  = (const float*)d_in[3];
    const float* bhh0  = (const float*)d_in[4];
    const float* Wih1  = (const float*)d_in[5];
    const float* Whh1  = (const float*)d_in[6];
    const float* bih1  = (const float*)d_in[7];
    const float* bhh1  = (const float*)d_in[8];
    const float* Wp    = (const float*)d_in[9];
    const float* bp    = (const float*)d_in[10];
    const float* gamma = (const float*)d_in[11];
    const float* beta  = (const float*)d_in[12];
    float* out = (float*)d_out;

    const int B = in_sizes[0] / (T_LEN * D_IN);   // 8192

    // allow >64KB dynamic LDS (WGP has 320KB); idempotent + deterministic
    (void)hipFuncSetAttribute((const void*)lstm_fused_kernel,
                              hipFuncAttributeMaxDynamicSharedMemorySize, SMEM_BYTES);

    lstm_fused_kernel<<<dim3(B / 16), dim3(128), SMEM_BYTES, stream>>>(
        x, Wih0, Whh0, bih0, bhh0, Wih1, Whh1, bih1, bhh1, Wp, bp, gamma, beta, out);

    (void)n_in; (void)out_size; (void)d_ws; (void)ws_size;
}
